// Att_75453985457442
// MI455X (gfx1250) — matrix operand order repaired and checked by reading, not compiled, not run on hardware
//
#include <hip/hip_runtime.h>
#include <hip/hip_bf16.h>

typedef __attribute__((ext_vector_type(16))) _Float16 v16h;
typedef __attribute__((ext_vector_type(8)))  float    v8f;
typedef __attribute__((ext_vector_type(4)))  unsigned int v4u;
typedef __attribute__((ext_vector_type(8)))  int          v8i;
typedef __attribute__((ext_vector_type(4)))  int          v4i;

#define DIM   96
#define NH    3
#define HD    32
#define WS    8
#define NTOK  64          // WS*WS
#define HW    65536       // 256*256
#define NWIN  33          // (256+8)/8 padded windows per side
#define NPIX  524288L     // 8 * 256 * 256

#ifndef __has_builtin
#define __has_builtin(x) 0
#endif
#if __has_builtin(__builtin_amdgcn_tensor_load_to_lds) && \
    __has_builtin(__builtin_amdgcn_s_wait_tensorcnt)
#define HAVE_TDM 1
#else
#define HAVE_TDM 0
#endif

__device__ __forceinline__ v8f wmma16(v16h a, v16h b, v8f c) {
  return __builtin_amdgcn_wmma_f32_16x16x32_f16(
      /*neg_a=*/false, a, /*neg_b=*/false, b,
      /*c_mod=*/(short)0, c, /*reuse_a=*/false, /*reuse_b=*/false);
}

// A-matrix (16x32 f16) lane/half -> K index  (contiguous order k = 16*laneHalf + hi, same as b_kmap)
__device__ __forceinline__ int a_kmap(int hi, int laneHalf) {
  return (laneHalf << 4) + hi;
}
// B-matrix (32x16 f16) lane/half -> K index
__device__ __forceinline__ int b_kmap(int hi, int laneHalf) {
  return (laneHalf << 4) + hi;
}

// reflect for SHIFT=4 pad into [0,256)
__device__ __forceinline__ int reflect_shift(int p) {
  int s = p - 4;
  s = (s < 0) ? -s : s;
  s = (s >= 256) ? 510 - s : s;
  return s;
}
// reflect for pad=2 (depthwise conv)
__device__ __forceinline__ int reflect2(int s) {
  s = (s < 0) ? -s : s;
  s = (s > 255) ? 510 - s : s;
  return s;
}

#if HAVE_TDM
// TDM: DMA one [ch=32][y=8][x=8] f32 tile of the [288,256,256] qkv tensor
// into LDS. D# packing per CDNA5 ISA 8.3-8.6 (data_size=4B, 3D tile).
__device__ __forceinline__ void tdm_load_tile(unsigned lds_addr,
                                              const float* gptr) {
  unsigned long long ga = (unsigned long long)gptr;
  v4u g0 = { 1u,                                  // count=1, user mode
             lds_addr,                            // LDS byte address
             (unsigned)(ga & 0xFFFFFFFFu),        // global_addr[31:0]
             (unsigned)((ga >> 32) & 0x1FFFFFFu)  // global_addr[56:32]
                 | (2u << 30) };                  // type=2 (image)
  v8i g1 = { (int)(2u << 16),                     // data_size=2 -> 4 bytes
             (int)(256u << 16),                   // tensor_dim0[15:0]=256
             (int)(256u << 16),                   // dim0 hi=0 | tensor_dim1 lo=256
             (int)(8u << 16),                     // dim1 hi=0 | tile_dim0=8
             (int)(8u | (32u << 16)),             // tile_dim1=8 | tile_dim2=32
             256,                                 // tensor_dim0_stride=256
             0,                                   // stride0 hi | stride1 lo (65536&0xffff=0)
             1 };                                 // tensor_dim1_stride[47:16]=1 (=65536)
  v4i g2 = { 288,                                 // tensor_dim2
             1,                                   // tensor_dim3 (unused)
             65536,                               // tensor_dim2_stride lo
             0 };                                 // stride2 hi | tile_dim3=0 (3D)
  v4i g3 = { 0, 0, 0, 0 };
#if __clang_major__ >= 23
  v8i g4 = { 0, 0, 0, 0, 0, 0, 0, 0 };
  __builtin_amdgcn_tensor_load_to_lds(g0, g1, g2, g3, g4, 0);
#else
  __builtin_amdgcn_tensor_load_to_lds(g0, g1, g2, g3, 0);
#endif
}
#endif

// ---------------------------------------------------------------------------
// Kernel 1: Swin-V2 continuous rel-pos bias MLP: [4096,2] -> relu 256 -> 3
// ---------------------------------------------------------------------------
__global__ __launch_bounds__(128) void bias_mlp(
    const float* __restrict__ m1_w, const float* __restrict__ m1_b,
    const float* __restrict__ m2_w, const float* __restrict__ m2_b,
    float* __restrict__ bias) {
  int p = blockIdx.x * 128 + threadIdx.x;      // 0..4095
  if (p >= NTOK * NTOK) return;
  int i = p >> 6, j = p & 63;
  float dy = (float)((i >> 3) - (j >> 3));
  float dx = (float)((i & 7) - (j & 7));
  float r0 = copysignf(log1pf(fabsf(dy)), dy);
  float r1 = copysignf(log1pf(fabsf(dx)), dx);
  float o0 = m2_b[0], o1 = m2_b[1], o2 = m2_b[2];
#pragma unroll 8
  for (int h = 0; h < 256; ++h) {
    float hv = fmaxf(fmaf(r0, m1_w[h], fmaf(r1, m1_w[256 + h], m1_b[h])), 0.0f);
    o0 = fmaf(hv, m2_w[h * 3 + 0], o0);
    o1 = fmaf(hv, m2_w[h * 3 + 1], o1);
    o2 = fmaf(hv, m2_w[h * 3 + 2], o2);
  }
  bias[0 * 4096 + p] = o0;
  bias[1 * 4096 + p] = o1;
  bias[2 * 4096 + p] = o2;
}

// ---------------------------------------------------------------------------
// Kernel 2: fused q/k/v 1x1 conv as WMMA GEMM:  [288 x 96] @ [96 x NPIX]
// grid = (18 M-tiles, 4096 N-blocks); block = 256 (8 waves x 16 pixels)
// N-major block order -> x streamed through L2 once across all M-tiles.
// ---------------------------------------------------------------------------
__global__ __launch_bounds__(256) void qkv_gemm(
    const float* __restrict__ x,
    const float* __restrict__ qk_w, const float* __restrict__ qk_b,
    const float* __restrict__ v_w,  const float* __restrict__ v_b,
    float* __restrict__ qkv) {
  __shared__ _Float16 wl[16][96];
  __shared__ float    bl[16];
  const int mt  = blockIdx.x;            // 0..17
  const int tid = threadIdx.x;

  for (int i = tid; i < 16 * 96; i += 256) {
    int row = i / 96, kk = i % 96;
    int oc = mt * 16 + row;
    float w = (oc < 192) ? qk_w[oc * 96 + kk] : v_w[(oc - 192) * 96 + kk];
    wl[row][kk] = (_Float16)w;
  }
  if (tid < 16) {
    int oc = mt * 16 + tid;
    bl[tid] = (oc < 192) ? qk_b[oc] : v_b[oc - 192];
  }
  __syncthreads();

  const int  wave = tid >> 5, lane = tid & 31;
  const int  lb = lane & 15, lh = lane >> 4;
  const long pix  = (long)blockIdx.y * 128 + wave * 16 + lb;
  const long bimg = pix >> 16;
  const long hw   = pix & 65535;

  __builtin_prefetch(x + (bimg * 96) * (long)HW + hw, 0, 1);

  v8f acc = {};
#pragma unroll
  for (int ks = 0; ks < 3; ++ks) {
    v16h a, b;
#pragma unroll
    for (int hi = 0; hi < 16; ++hi)
      a[hi] = wl[lb][ks * 32 + a_kmap(hi, lh)];
    const float* xp = x + (bimg * 96 + ks * 32 + (lh << 4)) * (long)HW + hw;
#pragma unroll
    for (int hi = 0; hi < 16; ++hi)
      b[hi] = (_Float16)xp[(long)hi * HW];
    acc = wmma16(a, b, acc);
  }
#pragma unroll
  for (int r = 0; r < 8; ++r) {
    int row = r + (lh << 3);
    int oc  = mt * 16 + row;
    qkv[(bimg * 288 + oc) * (long)HW + hw] = acc[r] + bl[row];
  }
}

// ---------------------------------------------------------------------------
// Kernel 3: shifted-window attention, one block per (window, head)
// grid = 8*33*33*3 blocks; block = 128 (4 waves, each owns a 16-row strip).
// Interior windows: Q/K/V tiles DMA'd to LDS by the Tensor Data Mover
// (3 rectangular 3D tiles), completion via s_wait_tensorcnt + barrier.
// Border windows: manual gather with fused reflect-pad + shift indexing.
// ---------------------------------------------------------------------------
__global__ __launch_bounds__(128) void win_attn(
    const float* __restrict__ qkv, const float* __restrict__ bias,
    float* __restrict__ attn_out) {
  __shared__ float    Qf[HD][NTOK];   // [ch][y*8+x]  (TDM tile order)
  __shared__ float    Kf[HD][NTOK];
  __shared__ float    Vf[HD][NTOK];
  __shared__ float    S[NTOK][NTOK];
  __shared__ _Float16 Ph[NTOK][NTOK];

  const int bid  = blockIdx.x;
  const int h    = bid % NH;
  const int wi   = bid / NH;
  const long bimg = wi / (NWIN * NWIN);
  const int r33  = wi % (NWIN * NWIN);
  const int wy = r33 / NWIN, wx = r33 % NWIN;
  const int tid = threadIdx.x;
  const float scale = 0.17677669529663687f;   // 32^-0.5

#if HAVE_TDM
  const bool interior = (wy >= 1) && (wy <= 31) && (wx >= 1) && (wx <= 31);
  if (interior) {
    if (tid < 32) {   // wave 0 owns the TENSORcnt for all three DMAs
      int gy0 = wy * WS - 4, gx0 = wx * WS - 4;
      const float* base =
          qkv + (bimg * 288) * (long)HW + (long)gy0 * 256 + gx0;
      tdm_load_tile((unsigned)(unsigned long long)(void*)&Qf[0][0],
                    base + (long)(h * HD) * HW);
      tdm_load_tile((unsigned)(unsigned long long)(void*)&Kf[0][0],
                    base + (long)(96 + h * HD) * HW);
      tdm_load_tile((unsigned)(unsigned long long)(void*)&Vf[0][0],
                    base + (long)(192 + h * HD) * HW);
      __builtin_amdgcn_s_wait_tensorcnt(0);
    }
  } else
#endif
  {
    for (int idx = tid; idx < HD * NTOK; idx += 128) {
      int d = idx >> 6, t = idx & 63;
      int gy = reflect_shift(wy * WS + (t >> 3));
      int gx = reflect_shift(wx * WS + (t & 7));
      long base = (bimg * 288) * (long)HW + (long)gy * 256 + gx;
      long cq = (long)(h * HD + d) * HW;
      Qf[d][t] = qkv[base + cq];
      Kf[d][t] = qkv[base + cq + 96L * HW];
      Vf[d][t] = qkv[base + cq + 192L * HW];
    }
  }
  __syncthreads();

  const int wave = tid >> 5, lane = tid & 31;
  const int lb = lane & 15, lh = lane >> 4;
  const int m0 = wave * 16;

  // S = (Q*scale) K^T + bias   (K-dim = 32 -> single WMMA per tile)
  {
    v16h a;
#pragma unroll
    for (int hi = 0; hi < 16; ++hi)
      a[hi] = (_Float16)(Qf[a_kmap(hi, lh)][m0 + lb] * scale);
#pragma unroll
    for (int nt = 0; nt < 4; ++nt) {
      v16h b;
#pragma unroll
      for (int hi = 0; hi < 16; ++hi)
        b[hi] = (_Float16)Kf[b_kmap(hi, lh)][nt * 16 + lb];
      v8f c = {};
      c = wmma16(a, b, c);
#pragma unroll
      for (int r = 0; r < 8; ++r) {
        int m = m0 + r + (lh << 3);
        int n = nt * 16 + lb;
        S[m][n] = c[r] + bias[h * 4096 + m * 64 + n];
      }
    }
  }
  __syncthreads();

  // row softmax (thread t owns row t)
  if (tid < NTOK) {
    float mx = -1e30f;
#pragma unroll 8
    for (int n = 0; n < NTOK; ++n) mx = fmaxf(mx, S[tid][n]);
    float sum = 0.0f;
#pragma unroll 8
    for (int n = 0; n < NTOK; ++n) {
      float e = __expf(S[tid][n] - mx);
      S[tid][n] = e;
      sum += e;
    }
    float inv = 1.0f / sum;
#pragma unroll 8
    for (int n = 0; n < NTOK; ++n) Ph[tid][n] = (_Float16)(S[tid][n] * inv);
  }
  __syncthreads();

  // O = P V  (K-dim = 64 -> 2 WMMA steps), fused window-reverse + crop store
#pragma unroll
  for (int nt = 0; nt < 2; ++nt) {
    v8f acc = {};
#pragma unroll
    for (int ks = 0; ks < 2; ++ks) {
      v16h a, b;
#pragma unroll
      for (int hi = 0; hi < 16; ++hi)
        a[hi] = Ph[m0 + lb][ks * 32 + a_kmap(hi, lh)];
#pragma unroll
      for (int hi = 0; hi < 16; ++hi)
        b[hi] = (_Float16)Vf[nt * 16 + lb][ks * 32 + (lh << 4) + hi];
      acc = wmma16(a, b, acc);
    }
#pragma unroll
    for (int r = 0; r < 8; ++r) {
      int m  = m0 + r + (lh << 3);
      int dd = nt * 16 + lb;
      int oy = wy * WS + (m >> 3) - 4;
      int ox = wx * WS + (m & 7) - 4;
      if (oy >= 0 && oy < 256 && ox >= 0 && ox < 256)
        attn_out[(bimg * 96 + h * HD + dd) * (long)HW + (long)oy * 256 + ox] =
            acc[r];
    }
  }
}

// ---------------------------------------------------------------------------
// Kernel 4: depthwise 5x5 conv on v (reflect pad) accumulated into attn_out
// ---------------------------------------------------------------------------
__global__ __launch_bounds__(256) void dwconv_add(
    const float* __restrict__ qkv, const float* __restrict__ conv_w,
    const float* __restrict__ conv_b, float* __restrict__ attn_out) {
  long idx = (long)blockIdx.x * 256 + threadIdx.x;   // [b][c][y][x]
  int xq = (int)(idx & 255);
  int yq = (int)((idx >> 8) & 255);
  long bc = idx >> 16;
  int  c  = (int)(bc % 96);
  long bimg = bc / 96;

  const float* vp = qkv + (bimg * 288 + 192 + c) * (long)HW;
  const float* wp = conv_w + c * 25;
  float acc = conv_b[c];
#pragma unroll
  for (int ky = 0; ky < 5; ++ky) {
    int yy = reflect2(yq + ky - 2);
#pragma unroll
    for (int kx = 0; kx < 5; ++kx) {
      int xx = reflect2(xq + kx - 2);
      acc = fmaf(vp[yy * 256 + xx], wp[ky * 5 + kx], acc);
    }
  }
  attn_out[idx] += acc;   // attn_out fully rewritten by win_attn each call
}

// ---------------------------------------------------------------------------
// Kernel 5: proj 1x1 conv as WMMA GEMM: [96 x 96] @ [96 x NPIX] -> d_out
// ---------------------------------------------------------------------------
__global__ __launch_bounds__(256) void proj_gemm(
    const float* __restrict__ in, const float* __restrict__ w,
    const float* __restrict__ bvec, float* __restrict__ out) {
  __shared__ _Float16 wl[16][96];
  __shared__ float    bl[16];
  const int mt  = blockIdx.x;            // 0..5
  const int tid = threadIdx.x;

  for (int i = tid; i < 16 * 96; i += 256) {
    int row = i / 96, kk = i % 96;
    wl[row][kk] = (_Float16)w[(mt * 16 + row) * 96 + kk];
  }
  if (tid < 16) bl[tid] = bvec[mt * 16 + tid];
  __syncthreads();

  const int  wave = tid >> 5, lane = tid & 31;
  const int  lb = lane & 15, lh = lane >> 4;
  const long pix  = (long)blockIdx.y * 128 + wave * 16 + lb;
  const long bimg = pix >> 16;
  const long hw   = pix & 65535;

  v8f acc = {};
#pragma unroll
  for (int ks = 0; ks < 3; ++ks) {
    v16h a, b;
#pragma unroll
    for (int hi = 0; hi < 16; ++hi)
      a[hi] = wl[lb][ks * 32 + a_kmap(hi, lh)];
    const float* ip = in + (bimg * 96 + ks * 32 + (lh << 4)) * (long)HW + hw;
#pragma unroll
    for (int hi = 0; hi < 16; ++hi)
      b[hi] = (_Float16)ip[(long)hi * HW];
    acc = wmma16(a, b, acc);
  }
#pragma unroll
  for (int r = 0; r < 8; ++r) {
    int row = r + (lh << 3);
    out[(bimg * 96 + mt * 16 + row) * (long)HW + hw] = acc[r] + bl[row];
  }
}

// ---------------------------------------------------------------------------
extern "C" void kernel_launch(void* const* d_in, const int* in_sizes, int n_in,
                              void* d_out, int out_size, void* d_ws, size_t ws_size,
                              hipStream_t stream) {
  const float* x      = (const float*)d_in[0];
  const float* v_w    = (const float*)d_in[1];
  const float* v_b    = (const float*)d_in[2];
  const float* qk_w   = (const float*)d_in[3];
  const float* qk_b   = (const float*)d_in[4];
  const float* proj_w = (const float*)d_in[5];
  const float* proj_b = (const float*)d_in[6];
  const float* conv_w = (const float*)d_in[7];
  const float* conv_b = (const float*)d_in[8];
  const float* m1_w   = (const float*)d_in[9];
  const float* m1_b   = (const float*)d_in[10];
  const float* m2_w   = (const float*)d_in[11];
  const float* m2_b   = (const float*)d_in[12];

  float* ws   = (float*)d_ws;
  float* bias = ws;                          // 3*4096 floats
  float* qkv  = ws + 16384;                  // 8*288*65536 floats
  float* attn = qkv + 8L * 288 * HW;         // 8*96*65536 floats

  bias_mlp<<<32, 128, 0, stream>>>(m1_w, m1_b, m2_w, m2_b, bias);
  qkv_gemm<<<dim3(18, 4096), 256, 0, stream>>>(x, qk_w, qk_b, v_w, v_b, qkv);
  win_attn<<<8 * NWIN * NWIN * NH, 128, 0, stream>>>(qkv, bias, attn);
  dwconv_add<<<(int)(8L * 96 * HW / 256), 256, 0, stream>>>(qkv, conv_w, conv_b, attn);
  proj_gemm<<<dim3(6, 4096), 256, 0, stream>>>(attn, proj_w, proj_b, (float*)d_out);
}
